// ptBEVnet_28862180229824
// MI455X (gfx1250) — matrix operand order repaired and checked by reading, not compiled, not run on hardware
//
#include <hip/hip_runtime.h>
#include <hip/hip_bf16.h>

typedef __attribute__((ext_vector_type(16))) _Float16 v16h;
typedef __attribute__((ext_vector_type(8)))  _Float16 v8h;
typedef __attribute__((ext_vector_type(8)))  float    v8f;

#define NPTS   100000
#define FEAT   7
#define GH     480
#define GW     360
#define GSEG   (GH * GW)          // 172800
#define NH     32
#define BNEPS  1e-5f

// ---------------------------------------------------------------------------
// float atomic max via integer punning (init must be -inf = 0xFF800000)
// ---------------------------------------------------------------------------
__device__ __forceinline__ void atomicMaxF(float* addr, float v) {
    if (v >= 0.0f) atomicMax((int*)addr, __float_as_int(v));
    else           atomicMin((unsigned int*)addr, __float_as_uint(v));
}

// ---------------------------------------------------------------------------
// WMMA B-fragment swizzle: weights stored fragment-major so each lane's
// 16 halves are one contiguous, 32-byte-aligned chunk:
//   frag[((nt*nkt + kt)*32 + lane)*16 + i]
// where n = nt*16 + (lane&15), k = kt*32 + (i>>3)*16 + (lane>>4)*8 + (i&7).
// ---------------------------------------------------------------------------
__device__ __forceinline__ size_t frag_idx(int k, int n, int nkt) {
    int nt = n >> 4, laneN = n & 15;
    int kt = k >> 5, kk = k & 31;
    int lane = (((kk >> 3) & 1) << 4) + laneN;
    int i = (kk & 7) + ((kk >> 4) << 3);
    return ((size_t)(nt * nkt + kt) * 32 + lane) * 16 + i;
}

// ---------------------------------------------------------------------------
// prep: fold BN (eval mode) into fragment-swizzled f16 weights + f32 biases
// ---------------------------------------------------------------------------
struct PrepIn {
    const float *l1w, *l1b, *g1, *be1, *m1, *v1;
    const float *l2w, *l2b, *g2, *be2, *m2, *v2;
    const float *l3w, *l3b, *g3, *be3, *m3, *v3;
    const float *l4w, *l4b;
    const float *lcw, *lcb;
    const float *g0, *be0, *m0, *v0;
};
struct PrepOut {
    _Float16 *w1, *w2, *w3, *w4, *wc;
    float *b1, *b2, *b3, *b4, *bc;
    float *s0, *sh0;
};

__device__ void fold_layer(const float* Wt, const float* bi,
                           const float* g, const float* be,
                           const float* m, const float* v,
                           int din, int dinp, int dout,
                           _Float16* wo, float* bo, bool hasBN,
                           int tid, int nth) {
    const int nkt = dinp / 32;
    for (int idx = tid; idx < dinp * dout; idx += nth) {
        int k = idx / dout, n = idx % dout;
        float s = hasBN ? g[n] * rsqrtf(v[n] + BNEPS) : 1.0f;
        float val = (k < din) ? Wt[k * dout + n] * s : 0.0f;
        wo[frag_idx(k, n, nkt)] = (_Float16)val;
    }
    for (int n = tid; n < dout; n += nth) {
        float s = hasBN ? g[n] * rsqrtf(v[n] + BNEPS) : 1.0f;
        bo[n] = hasBN ? (bi[n] - m[n]) * s + be[n] : bi[n];
    }
}

__global__ void prep_kernel(PrepIn in, PrepOut out) {
    int tid = threadIdx.x, nth = blockDim.x;
    fold_layer(in.l1w, in.l1b, in.g1, in.be1, in.m1, in.v1,   7,  32,  64, out.w1, out.b1, true,  tid, nth);
    fold_layer(in.l2w, in.l2b, in.g2, in.be2, in.m2, in.v2,  64,  64, 128, out.w2, out.b2, true,  tid, nth);
    fold_layer(in.l3w, in.l3b, in.g3, in.be3, in.m3, in.v3, 128, 128, 256, out.w3, out.b3, true,  tid, nth);
    fold_layer(in.l4w, in.l4b, 0, 0, 0, 0,                  256, 256, 512, out.w4, out.b4, false, tid, nth);
    fold_layer(in.lcw, in.lcb, 0, 0, 0, 0,                  512, 512,  32, out.wc, out.bc, false, tid, nth);
    for (int f = tid; f < FEAT; f += nth) {
        float s = in.g0[f] * rsqrtf(in.v0[f] + BNEPS);
        out.s0[f]  = s;
        out.sh0[f] = in.be0[f] - in.m0[f] * s;
    }
}

// ---------------------------------------------------------------------------
// init segmax to -inf
// ---------------------------------------------------------------------------
__global__ void init_segmax(float* p, size_t n) {
    size_t stride = (size_t)gridDim.x * blockDim.x;
    for (size_t i = (size_t)blockIdx.x * blockDim.x + threadIdx.x; i < n; i += stride)
        p[i] = -INFINITY;
}

// ---------------------------------------------------------------------------
// A-fragment: 16x32 f16 from LDS (row m = lane&15; K split 0-7/16-23 vs
// 8-15/24-31 across half-waves), as two aligned 16B vector loads.
// ---------------------------------------------------------------------------
__device__ __forceinline__ v16h load_a_lds(const _Float16* aIn, int ldin,
                                           int kt, int mrow, int grp) {
    const v8h* p0 = (const v8h*)(aIn + mrow * ldin + kt * 32 + grp * 8);
    const v8h* p1 = (const v8h*)(aIn + mrow * ldin + kt * 32 + 16 + grp * 8);
    v8h a0 = *p0, a1 = *p1;
    v16h a;
#pragma unroll
    for (int i = 0; i < 8; ++i) { a[i] = a0[i]; a[8 + i] = a1[i]; }
    return a;
}

// ---------------------------------------------------------------------------
// one 16xdout layer step for a wave: 16 points, WMMA f32_16x16x32_f16,
// B fragments are single contiguous 32B loads from the swizzled weights.
// ---------------------------------------------------------------------------
__device__ __forceinline__ void wave_layer(const _Float16* __restrict__ aIn, int ldin,
                                           const _Float16* __restrict__ wgt,
                                           const float* __restrict__ bias,
                                           _Float16* __restrict__ aOut, int ldout,
                                           int din, int dout, int lane) {
    const int mrow = lane & 15;
    const int grp  = lane >> 4;
    const int nkt  = din / 32;
    const v16h* wf = (const v16h*)wgt;
    for (int nt = 0; nt < dout / 16; ++nt) {
        v8f c = {};
        for (int kt = 0; kt < nkt; ++kt) {
            v16h a = load_a_lds(aIn, ldin, kt, mrow, grp);
            v16h b = wf[(size_t)(nt * nkt + kt) * 32 + lane];
            c = __builtin_amdgcn_wmma_f32_16x16x32_f16(false, a, false, b,
                                                       (short)0, c, false, false);
        }
        int n = nt * 16 + mrow;
        float bn = bias[n];
#pragma unroll
        for (int r = 0; r < 8; ++r) {
            int m = r + grp * 8;
            float v = c[r] + bn;
            v = v > 0.0f ? v : 0.0f;                 // ReLU (layers 1-3)
            aOut[m * ldout + n] = (_Float16)v;
        }
    }
}

// ---------------------------------------------------------------------------
// fused MLP: 64 points per workgroup (4 waves x 16 points), layer-4 result
// streamed as atomic-max into segmax[seg][512]
// ---------------------------------------------------------------------------
__global__ void __launch_bounds__(128)
mlp_kernel(const float* __restrict__ pt_fea, const int* __restrict__ xy_ind,
           const float* __restrict__ s0, const float* __restrict__ sh0,
           const _Float16* __restrict__ w1, const float* __restrict__ b1,
           const _Float16* __restrict__ w2, const float* __restrict__ b2,
           const _Float16* __restrict__ w3, const float* __restrict__ b3,
           const _Float16* __restrict__ w4, const float* __restrict__ b4,
           float* __restrict__ segmax, int batch) {
    __shared__ __align__(16) _Float16 sA[64 * 128];   // input(ld32) then h2(ld128)
    __shared__ __align__(16) _Float16 sB[64 * 256];   // h1(ld64)   then h3(ld256)
    __shared__ int sSeg[64];

    const int tid    = threadIdx.x;
    const int lane   = tid & 31;
    const int wid    = tid >> 5;
    const int ptBase = blockIdx.x * 64;
    const size_t bN  = (size_t)batch * NPTS;

    // load 64 points, apply folded BN0, pad K 7 -> 32
    for (int idx = tid; idx < 64 * 32; idx += blockDim.x) {
        int p = idx >> 5, k = idx & 31;
        int pg = ptBase + p;
        float v = 0.0f;
        if (k < FEAT && pg < NPTS)
            v = pt_fea[(bN + pg) * FEAT + k] * s0[k] + sh0[k];
        sA[p * 32 + k] = (_Float16)v;
    }
    if (tid < 64) {
        int pg = ptBase + tid;
        int seg = 0;
        if (pg < NPTS) {
            int i0 = xy_ind[(bN + pg) * 2 + 0];
            int i1 = xy_ind[(bN + pg) * 2 + 1];
            seg = i0 * GW + i1;
        }
        sSeg[tid] = seg;
    }
    __syncthreads();

    // each wave computes its own 16 rows with no barrier, though wave 0's h2/h3 overlap other waves' input/h1
    wave_layer(sA + wid * 16 * 32,  32,  w1, b1, sB + wid * 16 * 64,  64,   32,  64, lane);
    wave_layer(sB + wid * 16 * 64,  64,  w2, b2, sA + wid * 16 * 128, 128,  64, 128, lane);
    wave_layer(sA + wid * 16 * 128, 128, w3, b3, sB + wid * 16 * 256, 256, 128, 256, lane);

    // layer 4: 256 -> 512, no ReLU/BN; atomic-max into segmax
    const _Float16* aIn = sB + wid * 16 * 256;
    const int mrow = lane & 15;
    const int grp  = lane >> 4;
    const v16h* wf4 = (const v16h*)w4;
    for (int nt = 0; nt < 32; ++nt) {
        v8f c = {};
        for (int kt = 0; kt < 8; ++kt) {
            v16h a = load_a_lds(aIn, 256, kt, mrow, grp);
            v16h b = wf4[(size_t)(nt * 8 + kt) * 32 + lane];
            c = __builtin_amdgcn_wmma_f32_16x16x32_f16(false, a, false, b,
                                                       (short)0, c, false, false);
        }
        int n = nt * 16 + mrow;
        float bn = b4[n];
#pragma unroll
        for (int r = 0; r < 8; ++r) {
            int m  = r + grp * 8;
            int pg = ptBase + wid * 16 + m;
            if (pg < NPTS) {
                int seg = sSeg[wid * 16 + m];
                atomicMaxF(&segmax[(size_t)seg * 512 + n], c[r] + bn);
            }
        }
    }
}

// ---------------------------------------------------------------------------
// compression: [G,512] segmax @ [512,32] + bias, ReLU, present mask,
// scatter to out[c*G + seg]. One wave per 16-segment tile.
// ---------------------------------------------------------------------------
__global__ void __launch_bounds__(128)
compress_kernel(const float* __restrict__ segmax,
                const _Float16* __restrict__ wc, const float* __restrict__ bc,
                float* __restrict__ out) {
    const int lane = threadIdx.x & 31;
    const int wid  = threadIdx.x >> 5;
    const int tile = blockIdx.x * 4 + wid;
    if (tile >= GSEG / 16) return;
    const int seg0 = tile * 16;
    const int mrow = lane & 15;
    const int grp  = lane >> 4;
    const v16h* wf = (const v16h*)wc;    // nkt = 16, 2 n-tiles

    v8f c0 = {}, c1 = {};
    const float* arow = segmax + (size_t)(seg0 + mrow) * 512;
    for (int kt = 0; kt < 16; ++kt) {
        v16h a;
#pragma unroll
        for (int i = 0; i < 8; ++i) {
            a[i]     = (_Float16)arow[kt * 32 + grp * 8 + i];
            a[8 + i] = (_Float16)arow[kt * 32 + 16 + grp * 8 + i];
        }
        v16h b0 = wf[(size_t)(0 * 16 + kt) * 32 + lane];
        v16h b1 = wf[(size_t)(1 * 16 + kt) * 32 + lane];
        c0 = __builtin_amdgcn_wmma_f32_16x16x32_f16(false, a, false, b0, (short)0, c0, false, false);
        c1 = __builtin_amdgcn_wmma_f32_16x16x32_f16(false, a, false, b1, (short)0, c1, false, false);
    }
#pragma unroll
    for (int r = 0; r < 8; ++r) {
        int m   = r + grp * 8;
        int seg = seg0 + m;
        bool present = segmax[(size_t)seg * 512] > -INFINITY;
        int n0 = mrow, n1 = 16 + mrow;
        float v0 = present ? fmaxf(c0[r] + bc[n0], 0.0f) : 0.0f;
        float v1 = present ? fmaxf(c1[r] + bc[n1], 0.0f) : 0.0f;
        out[(size_t)n0 * GSEG + seg] = v0;
        out[(size_t)n1 * GSEG + seg] = v1;
    }
}

// ---------------------------------------------------------------------------
// launch
// ---------------------------------------------------------------------------
extern "C" void kernel_launch(void* const* d_in, const int* in_sizes, int n_in,
                              void* d_out, int out_size, void* d_ws, size_t ws_size,
                              hipStream_t stream) {
    const float* pt_fea = (const float*)d_in[0];
    const int*   xy_ind = (const int*)d_in[1];

    PrepIn pin;
    pin.g0 = (const float*)d_in[2];  pin.be0 = (const float*)d_in[3];
    pin.m0 = (const float*)d_in[4];  pin.v0  = (const float*)d_in[5];
    pin.l1w = (const float*)d_in[6]; pin.l1b = (const float*)d_in[7];
    pin.g1 = (const float*)d_in[8];  pin.be1 = (const float*)d_in[9];
    pin.m1 = (const float*)d_in[10]; pin.v1  = (const float*)d_in[11];
    pin.l2w = (const float*)d_in[12]; pin.l2b = (const float*)d_in[13];
    pin.g2 = (const float*)d_in[14]; pin.be2 = (const float*)d_in[15];
    pin.m2 = (const float*)d_in[16]; pin.v2  = (const float*)d_in[17];
    pin.l3w = (const float*)d_in[18]; pin.l3b = (const float*)d_in[19];
    pin.g3 = (const float*)d_in[20]; pin.be3 = (const float*)d_in[21];
    pin.m3 = (const float*)d_in[22]; pin.v3  = (const float*)d_in[23];
    pin.l4w = (const float*)d_in[24]; pin.l4b = (const float*)d_in[25];
    pin.lcw = (const float*)d_in[26]; pin.lcb = (const float*)d_in[27];

    // workspace layout: segmax f32 [G*512] | swizzled f16 weights | f32 biases
    const size_t segElems = (size_t)GSEG * 512;
    float*     segmax = (float*)d_ws;
    _Float16*  hbase  = (_Float16*)((char*)d_ws + segElems * sizeof(float));
    PrepOut pout;
    pout.w1 = hbase;                     // 32*64
    pout.w2 = pout.w1 + 32 * 64;         // 64*128
    pout.w3 = pout.w2 + 64 * 128;        // 128*256
    pout.w4 = pout.w3 + 128 * 256;       // 256*512
    pout.wc = pout.w4 + 256 * 512;       // 512*32
    float* fbase = (float*)(pout.wc + 512 * 32);
    pout.b1 = fbase;            pout.b2 = pout.b1 + 64;
    pout.b3 = pout.b2 + 128;    pout.b4 = pout.b3 + 256;
    pout.bc = pout.b4 + 512;
    pout.s0 = pout.bc + 32;     pout.sh0 = pout.s0 + FEAT;

    prep_kernel<<<1, 256, 0, stream>>>(pin, pout);

    const int mlpBlocks = (NPTS + 63) / 64;          // 1563
    const int cmpBlocks = (GSEG / 16 + 3) / 4;       // 2700
    for (int b = 0; b < 2; ++b) {
        init_segmax<<<4096, 256, 0, stream>>>(segmax, segElems);
        mlp_kernel<<<mlpBlocks, 128, 0, stream>>>(
            pt_fea, xy_ind, pout.s0, pout.sh0,
            pout.w1, pout.b1, pout.w2, pout.b2,
            pout.w3, pout.b3, pout.w4, pout.b4,
            segmax, b);
        compress_kernel<<<cmpBlocks, 128, 0, stream>>>(
            segmax, pout.wc, pout.bc,
            (float*)d_out + (size_t)b * NH * GSEG);
    }
}
